// DeepGCN_16071767622287
// MI455X (gfx1250) — matrix operand order repaired and checked by reading, not compiled, not run on hardware
//
#include <hip/hip_runtime.h>
#include <hip/hip_bf16.h>

// ---------------------------------------------------------------------------
// DeepGCN forward on MI455X (gfx1250, wave32)
//   GEMMs: bf16 WMMA 16x16x32, fp32 accumulate. K templated (full unroll,
//          immediate-offset addressing). Weights pre-packed into WMMA-B
//          fragment layout (one contiguous 32B load per lane). A tiles staged
//          bf16 through LDS with b64 load/store per thread.
//   SpMM : per-edge coalesced gather + global_atomic_add_f32 scatter; the
//          40000x128 accumulator lives in the 192MB L2, so atomics run at L2.
// ---------------------------------------------------------------------------

typedef __attribute__((ext_vector_type(16))) __bf16 v16bf;
typedef __attribute__((ext_vector_type(8)))  float  v8f;

#define NNODES 40000
#define NEDGES 640000
#define DIN    256
#define HDIM   128
#define CDIM   64

// Pack fp32 row-major W[K][Nc] into WMMA-B bf16 fragments:
//   Wp[((kt*Nc + n)*2 + hi)*16 + j] = bf16( W[(kt*32 + hi*16 + j)*Nc + n] )
__global__ __launch_bounds__(256)
void pack_w_bf16(const float* __restrict__ W, __bf16* __restrict__ Wp,
                 int K, int Nc) {
  int p = blockIdx.x * blockDim.x + threadIdx.x;
  if (p >= K * Nc) return;
  int j  = p & 15;
  int q  = p >> 4;
  int hi = q & 1; q >>= 1;
  int n  = q % Nc;
  int kt = q / Nc;
  int k  = kt * 32 + hi * 16 + j;
  Wp[p] = (__bf16)W[(size_t)k * Nc + n];
}

__global__ __launch_bounds__(256)
void pack_bf16(const float* __restrict__ src, __bf16* __restrict__ dst, int n) {
  int i = blockIdx.x * blockDim.x + threadIdx.x;
  if (i < n) dst[i] = (__bf16)src[i];
}

// Y[M,Nc] = Xb[M,K](bf16) @ W + bias.  Block = 256 thr (8 waves), 32-row strip.
// NT = Nc/16.  NT==8: wave w owns col-tile w for BOTH 16-row M-tiles (2 WMMAs
// per k-step, shared B).  NT==4: wave w owns (mtile = w>>2, ntile = w&3).
template <int NT, int K>
__global__ __launch_bounds__(256)
void gemm_bias_wmma(const __bf16* __restrict__ Xb, const __bf16* __restrict__ Wp,
                    const float* __restrict__ bias, float* __restrict__ Y) {
  constexpr int Nc = NT * 16;
  constexpr int MT = (NT == 8) ? 2 : 1;
  constexpr int Kw = K >> 1;             // u32 words per row
  constexpr int KT = K >> 5;             // 32-wide k-tiles (8 or 4)
  __shared__ unsigned As32[32][18];      // 72B row stride: b64-aligned,
                                         // 18*ml mod 32 -> 16 distinct banks

  const int lane  = threadIdx.x & 31;
  const int wave  = threadIdx.x >> 5;
  const int hi    = lane >> 4;
  const int ml    = lane & 15;
  const int m0    = blockIdx.x << 5;     // 32-row strip (40000/32 = 1250 exact)
  const int ntile = (NT == 8) ? wave : (wave & 3);
  const int mbase = (NT == 8) ? 0 : (wave >> 2);
  const int n     = ntile * 16 + ml;

  // staging coords: thread -> (row, 2-u32 chunk); 512 u32 / 256 thr = 2 each
  const int sr = threadIdx.x >> 3;       // 0..31
  const int sc = (threadIdx.x & 7) << 1; // 0,2,..,14

  const unsigned* Xb32 = (const unsigned*)Xb;
  const unsigned* srcRow = Xb32 + (size_t)(m0 + sr) * Kw;

  const v8f vzero = {0.f, 0.f, 0.f, 0.f, 0.f, 0.f, 0.f, 0.f};
  v8f acc[MT];
#pragma unroll
  for (int mi = 0; mi < MT; ++mi) acc[mi] = vzero;

#pragma unroll
  for (int kt = 0; kt < KT; ++kt) {
    // stage 32x32 bf16 A tile: one b64 global load + one b64 LDS store / thread
    *(uint2*)&As32[sr][sc] = *(const uint2*)&srcRow[kt * 16 + sc];
    __syncthreads();

    // B fragment: one contiguous 32B load per lane from packed weights
    v16bf b = *(const v16bf*)(Wp + (((size_t)kt * Nc + n) * 2 + hi) * 16);

#pragma unroll
    for (int mi = 0; mi < MT; ++mi) {
      const int row = ((NT == 8) ? mi : mbase) * 16 + ml;
      // A fragment (contiguous k order, matching packed B): lane half hi gets
      // K = hi*16 + {0..15} (v0..7)
      union { v16bf v; unsigned u[8]; } af;
#pragma unroll
      for (int jj = 0; jj < 4; ++jj) {
        af.u[jj]     = As32[row][hi * 8 + jj];
        af.u[jj + 4] = As32[row][hi * 8 + 4 + jj];
      }
      acc[mi] = __builtin_amdgcn_wmma_f32_16x16x32_bf16(
          false, af.v, false, b, (short)0, acc[mi], false, false);
    }
    __syncthreads();
  }

  const float bv = bias[n];
#pragma unroll
  for (int mi = 0; mi < MT; ++mi) {
    const int mt = (NT == 8) ? mi : mbase;
#pragma unroll
    for (int r = 0; r < 8; ++r) {
      int m = mt * 16 + hi * 8 + r;        // C/D layout: acc[r] -> m = hi*8+r
      Y[(size_t)(m0 + m) * Nc + n] = acc[mi][r] + bv;
    }
  }
}

// out[row[e], :] += val[e] * Ysrc[col[e], :]
// One thread = 4 cols of one edge; a wave covers a full 128-col row
// (coalesced 512B gather). Scatter via global_atomic_add_f32 into L2.
__global__ __launch_bounds__(256)
void spmm_scatter(const int* __restrict__ row, const int* __restrict__ col,
                  const float* __restrict__ val, const float* __restrict__ Ysrc,
                  float* __restrict__ out, int nEdges, int D, int lgGroups) {
  long long g = (long long)blockIdx.x * blockDim.x + threadIdx.x;
  int e = (int)(g >> lgGroups);
  if (e >= nEdges) return;
  int c = ((int)g & ((1 << lgGroups) - 1)) << 2;

  float v = val[e];
  size_t yoff = (size_t)col[e] * D + c;
  size_t ooff = (size_t)row[e] * D + c;
  float4 y = *(const float4*)(Ysrc + yoff);
  atomicAdd(out + ooff + 0, v * y.x);
  atomicAdd(out + ooff + 1, v * y.y);
  atomicAdd(out + ooff + 2, v * y.z);
  atomicAdd(out + ooff + 3, v * y.w);
}

// h = relu(s) in fp32, plus bf16 mirror for the next WMMA GEMM
__global__ __launch_bounds__(256)
void relu_copy2(float* __restrict__ h, __bf16* __restrict__ hb,
                const float* __restrict__ s, int n) {
  int i = blockIdx.x * blockDim.x + threadIdx.x;
  if (i < n) {
    float v = s[i];
    v = v > 0.f ? v : 0.f;
    h[i]  = v;
    hb[i] = (__bf16)v;
  }
}

// h += relu(s)*dt in fp32, plus bf16 mirror
__global__ __launch_bounds__(256)
void residual_relu2(float* __restrict__ h, __bf16* __restrict__ hb,
                    const float* __restrict__ s, const float* __restrict__ dt,
                    int n) {
  int i = blockIdx.x * blockDim.x + threadIdx.x;
  if (i < n) {
    float v = s[i];
    v = v > 0.f ? v : 0.f;
    float nh = h[i] + v * dt[0];
    h[i]  = nh;
    hb[i] = (__bf16)nh;
  }
}

extern "C" void kernel_launch(void* const* d_in, const int* in_sizes, int n_in,
                              void* d_out, int out_size, void* d_ws, size_t ws_size,
                              hipStream_t stream) {
  const float* x     = (const float*)d_in[0];
  const int*   erow  = (const int*)  d_in[1];
  const int*   ecol  = (const int*)  d_in[2];
  const float* eval_ = (const float*)d_in[3];
  const float* w1    = (const float*)d_in[4];
  const float* b1    = (const float*)d_in[5];
  const float* wm    = (const float*)d_in[6];
  const float* bm    = (const float*)d_in[7];
  const float* w2    = (const float*)d_in[8];
  const float* b2    = (const float*)d_in[9];
  const float* dt    = (const float*)d_in[10];
  float* out = (float*)d_out;

  const int elemH = NNODES * HDIM;          // 5.12M
  const int elemX = NNODES * DIN;           // 10.24M

  // workspace layout (all offsets 32B-aligned):
  //   h (N*H f32) | t (N*H f32) | s (N*H f32, aliased as xb bf16) |
  //   hb (N*H bf16) | packed weights
  float*  h  = (float*)d_ws;
  float*  t  = h + (size_t)elemH;
  float*  s  = t + (size_t)elemH;
  __bf16* xb = (__bf16*)s;                  // N*DIN bf16 == N*H f32 bytes; dead
                                            // before s's first memset
  __bf16* hb   = (__bf16*)(s + (size_t)elemH);
  __bf16* w1p  = hb + (size_t)elemH;
  __bf16* wm0p = w1p + DIN * HDIM;
  __bf16* wm1p = wm0p + HDIM * HDIM;
  __bf16* w2p  = wm1p + HDIM * HDIM;

  const int gBlocks = NNODES / 32;          // 1250 exact
  dim3 b256(256);

  // ---- one-time packs (weights are L2-resident; x -> bf16) ----
  pack_bf16<<<(elemX + 255) / 256, b256, 0, stream>>>(x, xb, elemX);
  pack_w_bf16<<<(DIN * HDIM + 255) / 256, b256, 0, stream>>>(w1, w1p, DIN, HDIM);
  pack_w_bf16<<<(HDIM * HDIM + 255) / 256, b256, 0, stream>>>(wm, wm0p, HDIM, HDIM);
  pack_w_bf16<<<(HDIM * HDIM + 255) / 256, b256, 0, stream>>>(
      wm + (size_t)HDIM * HDIM, wm1p, HDIM, HDIM);
  pack_w_bf16<<<(HDIM * CDIM + 255) / 256, b256, 0, stream>>>(w2, w2p, HDIM, CDIM);

  // ---- layer 1: h = relu(A @ (x @ w1 + b1)) ----
  gemm_bias_wmma<8, DIN><<<gBlocks, b256, 0, stream>>>(xb, w1p, b1, t);
  hipMemsetAsync(s, 0, sizeof(float) * (size_t)elemH, stream);  // kills xb (dead)
  {
    long long th = (long long)NEDGES * (HDIM / 4);
    spmm_scatter<<<(int)((th + 255) / 256), b256, 0, stream>>>(
        erow, ecol, eval_, t, s, NEDGES, HDIM, 5);
  }
  relu_copy2<<<(elemH + 255) / 256, b256, 0, stream>>>(h, hb, s, elemH);

  // ---- middle residual layers: h += relu(A @ (h @ wm[i] + bm[i])) * dt ----
  const __bf16* wmp[2] = {wm0p, wm1p};
  for (int i = 0; i < 2; ++i) {
    gemm_bias_wmma<8, HDIM><<<gBlocks, b256, 0, stream>>>(
        hb, wmp[i], bm + (size_t)i * HDIM, t);
    hipMemsetAsync(s, 0, sizeof(float) * (size_t)elemH, stream);
    long long th = (long long)NEDGES * (HDIM / 4);
    spmm_scatter<<<(int)((th + 255) / 256), b256, 0, stream>>>(
        erow, ecol, eval_, t, s, NEDGES, HDIM, 5);
    residual_relu2<<<(elemH + 255) / 256, b256, 0, stream>>>(h, hb, s, dt, elemH);
  }

  // ---- output layer: out = A @ (h @ w2 + b2) ----
  gemm_bias_wmma<4, HDIM><<<gBlocks, b256, 0, stream>>>(hb, w2p, b2, t);
  hipMemsetAsync(out, 0, sizeof(float) * (size_t)NNODES * CDIM, stream);
  {
    long long th = (long long)NEDGES * (CDIM / 4);
    spmm_scatter<<<(int)((th + 255) / 256), b256, 0, stream>>>(
        erow, ecol, eval_, t, out, NEDGES, CDIM, 4);
  }
}
